// _FastAttention_23725399343801
// MI455X (gfx1250) — matrix operand order repaired and checked by reading, not compiled, not run on hardware
//
#include <hip/hip_runtime.h>
#include <hip/hip_bf16.h>

#define HEADS      16
#define DIM_HEAD   64
#define BATCH      4
#define SEQ        4096
#define INPUT_DIM  1024
#define INNER      1024
#define TOKENS     (BATCH * SEQ)     // 16384
#define QKV_COLS   (3 * INNER)       // 3072
#define SCALE_F    0.125f            // 64^-0.5

typedef __attribute__((ext_vector_type(16))) __bf16 v16bf;
typedef __attribute__((ext_vector_type(8)))  float  v8f;

__device__ __forceinline__ unsigned short f2bf(float f) {
    unsigned u = __float_as_uint(f);
    u += 0x7FFFu + ((u >> 16) & 1u);          // round-to-nearest-even
    return (unsigned short)(u >> 16);
}
__device__ __forceinline__ float bf2f(unsigned short h) {
    return __uint_as_float(((unsigned)h) << 16);
}

union FragCast { uint4 u4[2]; v16bf v; };

// 16-bit A-matrix 16x32 layout: lane m=L&15, half=L>>4:
//   VGPR0-3: K = 8*half + 0..7 ; VGPR4-7: K = 16 + 8*half + 0..7
__device__ __forceinline__ v16bf load_fragA(const unsigned short* base, int row0,
                                            int ld, int k0, int lane) {
    int m = lane & 15, half = lane >> 4;
    const unsigned short* p = base + (size_t)(row0 + m) * ld + k0 + 8 * half;
    FragCast t;
    t.u4[0] = *(const uint4*)(p);
    t.u4[1] = *(const uint4*)(p + 16);
    return t.v;
}
// 16-bit B-matrix 32x16 layout: lane n=L&15, half=L>>4: K = 8*half + 0..7, 16 + 8*half + 0..7
__device__ __forceinline__ v16bf load_fragB(const unsigned short* base, int col0,
                                            int ld, int k0, int lane) {
    int n = lane & 15, half = lane >> 4;
    const unsigned short* p = base + (size_t)(col0 + n) * ld + k0 + 8 * half;
    FragCast t;
    t.u4[0] = *(const uint4*)(p);
    t.u4[1] = *(const uint4*)(p + 16);
    return t.v;
}

// ---------------------------------------------------------------------------
// Async global->LDS copy of one 128x32 bf16 tile (row-major, K contiguous).
// 512 chunks of 16B; 256 threads issue 2 GLOBAL_LOAD_ASYNC_TO_LDS_B128 each
// (16B per lane per instruction). Tracked by ASYNCcnt.
// ---------------------------------------------------------------------------
#define LDT 40   // padded LDS row stride (ushorts): 80B rows keep 16B alignment

__device__ __forceinline__ void issue_tile_copy(unsigned lds_base,
                                                const unsigned short* g,
                                                int ldg, int tid) {
    #pragma unroll
    for (int i = 0; i < 2; ++i) {
        int chunk = tid + 256 * i;          // [0, 512)
        int row = chunk >> 2;               // [0, 128)
        int seg = chunk & 3;                // 4 x 8 ushorts = 32 (BK)
        unsigned loff = lds_base + (unsigned)((row * LDT + seg * 8) * 2);
        const void* gp = g + (size_t)row * ldg + seg * 8;
        asm volatile("global_load_async_to_lds_b128 %0, %1, off"
                     :: "v"(loff), "v"(gp) : "memory");
    }
}

// ---------------------------------------------------------------------------
// GEMM: C[M,N] = A[M,K] @ Bt[N,K]^T (+ bias). A, Bt bf16 row-major, K contig.
// 128x128 block tile, BK=32, double-buffered async global->LDS staging.
// 8 waves: 4 over M (32 rows), 2 over N (64 cols) -> 8 WMMA accumulators.
// ---------------------------------------------------------------------------
template<bool OUT_BF16>
__global__ __launch_bounds__(256) void gemm_wmma_async(
    const unsigned short* __restrict__ A,    // [M][K] bf16
    const unsigned short* __restrict__ Bt,   // [N][K] bf16
    const float* __restrict__ bias,
    void* __restrict__ Cout, int M, int N, int K)
{
    constexpr int BM = 128, BN = 128, BK = 32;
    __shared__ __align__(16) unsigned short sA[2][BM * LDT];
    __shared__ __align__(16) unsigned short sB[2][BN * LDT];

    const int tid  = threadIdx.x;
    const int lane = tid & 31;
    const int wave = tid >> 5;
    const int wm   = wave & 3;        // 4 waves along M
    const int wn   = wave >> 2;       // 2 waves along N
    const int m0   = blockIdx.y * BM;
    const int n0   = blockIdx.x * BN;

    const unsigned short* Atile = A  + (size_t)m0 * K;
    const unsigned short* Btile = Bt + (size_t)n0 * K;
    const unsigned offA[2] = { (unsigned)(size_t)&sA[0][0], (unsigned)(size_t)&sA[1][0] };
    const unsigned offB[2] = { (unsigned)(size_t)&sB[0][0], (unsigned)(size_t)&sB[1][0] };

    v8f z = {0.f,0.f,0.f,0.f,0.f,0.f,0.f,0.f};
    v8f acc[2][4];
    #pragma unroll
    for (int t = 0; t < 2; ++t)
        #pragma unroll
        for (int u = 0; u < 4; ++u) acc[t][u] = z;

    const int T = K / BK;
    issue_tile_copy(offA[0], Atile, K, tid);
    issue_tile_copy(offB[0], Btile, K, tid);

    for (int i = 0; i < T; ++i) {
        const int buf = i & 1;
        if (i + 1 < T) {
            issue_tile_copy(offA[buf ^ 1], Atile + (size_t)(i + 1) * BK, K, tid);
            issue_tile_copy(offB[buf ^ 1], Btile + (size_t)(i + 1) * BK, K, tid);
            asm volatile("s_wait_asynccnt 0x4" ::: "memory");  // tile i drained, i+1 in flight
        } else {
            asm volatile("s_wait_asynccnt 0x0" ::: "memory");
        }
        __syncthreads();

        v16bf afrag[2], bfrag[4];
        #pragma unroll
        for (int t = 0; t < 2; ++t)
            afrag[t] = load_fragA(&sA[buf][0], wm * 32 + t * 16, LDT, 0, lane);
        #pragma unroll
        for (int u = 0; u < 4; ++u)
            bfrag[u] = load_fragB(&sB[buf][0], wn * 64 + u * 16, LDT, 0, lane);

        #pragma unroll
        for (int t = 0; t < 2; ++t)
            #pragma unroll
            for (int u = 0; u < 4; ++u)
                acc[t][u] = __builtin_amdgcn_wmma_f32_16x16x32_bf16(
                    false, afrag[t], false, bfrag[u], (short)0, acc[t][u], false, false);
        __syncthreads();   // all waves done reading buf before it is overwritten
    }

    // C/D layout: lane: N = lane&15 ; rows = (lane>>4)*8 + vgpr
    unsigned short* Cb = (unsigned short*)Cout;
    float*          Cf = (float*)Cout;
    const int cn = lane & 15;
    const int rb = (lane >> 4) * 8;
    #pragma unroll
    for (int t = 0; t < 2; ++t)
        #pragma unroll
        for (int u = 0; u < 4; ++u) {
            int col = n0 + wn * 64 + u * 16 + cn;
            #pragma unroll
            for (int v = 0; v < 8; ++v) {
                int row = m0 + wm * 32 + t * 16 + rb + v;
                float val = acc[t][u][v];
                if (OUT_BF16) Cb[(size_t)row * N + col] = f2bf(val);
                else          Cf[(size_t)row * N + col] = val + bias[col];
            }
        }
}

// ---------------------------------------------------------------------------
// Prep: f32 -> bf16 elementwise (x), and f32 [R][C] -> bf16 [C][R] (weights).
// ---------------------------------------------------------------------------
__global__ __launch_bounds__(256) void convert_bf16_kernel(
    const float* __restrict__ in, unsigned short* __restrict__ out, unsigned n4)
{
    unsigned i = blockIdx.x * 256 + threadIdx.x;
    if (i < n4) {
        float4 v = *(const float4*)(in + (size_t)i * 4);
        size_t o = (size_t)i * 4;
        out[o + 0] = f2bf(v.x); out[o + 1] = f2bf(v.y);
        out[o + 2] = f2bf(v.z); out[o + 3] = f2bf(v.w);
    }
}

__global__ __launch_bounds__(256) void transpose_bf16_kernel(
    const float* __restrict__ in, unsigned short* __restrict__ out, int R, int C)
{
    __shared__ unsigned short tile[64][65];
    const int r0 = blockIdx.y * 64, c0 = blockIdx.x * 64;
    for (int i = threadIdx.x; i < 64 * 64; i += 256) {
        int r = i >> 6, c = i & 63;                       // coalesced read along c
        tile[r][c] = f2bf(in[(size_t)(r0 + r) * C + c0 + c]);
    }
    __syncthreads();
    for (int i = threadIdx.x; i < 64 * 64; i += 256) {
        int c = i >> 6, r = i & 63;                       // coalesced write along r
        out[(size_t)(c0 + c) * R + r0 + r] = tile[r][c];
    }
}

// ---------------------------------------------------------------------------
// Per-(b,h) global_q / global_k softmax reductions. One block per (b,h).
// ---------------------------------------------------------------------------
__global__ __launch_bounds__(256) void global_qk_kernel(
    const unsigned short* __restrict__ qkv,   // [TOKENS, 3072] bf16
    const float* __restrict__ wq, const float* __restrict__ wk,
    float* __restrict__ gq_out, float* __restrict__ gk_out)
{
    __shared__ float sl[SEQ];      // logits -> probs (16 KB)
    __shared__ float red[256];
    __shared__ float svec[64];
    __shared__ float sg[64];
    __shared__ float sscal;

    const int tid = threadIdx.x;
    const int bh  = blockIdx.x;
    const int b   = bh >> 4;
    const int h   = bh & 15;

    for (int pass = 0; pass < 2; ++pass) {
        if (tid < 64) {
            float c = (pass == 0) ? wq[h * 64 + tid] * SCALE_F
                                  : sg[tid] * wk[h * 64 + tid] * SCALE_F;
            svec[tid] = c;
        }
        __syncthreads();
        const size_t colOff = (pass == 0) ? (size_t)(h * 64) : (size_t)(INNER + h * 64);

        float lmax = -1e30f;
        for (int n = tid; n < SEQ; n += 256) {
            const unsigned short* row = qkv + (size_t)(b * SEQ + n) * QKV_COLS + colOff;
            float dot = 0.f;
            #pragma unroll 8
            for (int d = 0; d < 64; ++d) dot += bf2f(row[d]) * svec[d];
            sl[n] = dot;
            lmax = fmaxf(lmax, dot);
        }
        red[tid] = lmax; __syncthreads();
        for (int s = 128; s > 0; s >>= 1) {
            if (tid < s) red[tid] = fmaxf(red[tid], red[tid + s]);
            __syncthreads();
        }
        lmax = red[0];
        __syncthreads();

        float lsum = 0.f;
        for (int n = tid; n < SEQ; n += 256) {
            float p = __expf(sl[n] - lmax);
            sl[n] = p; lsum += p;
        }
        red[tid] = lsum; __syncthreads();
        for (int s = 128; s > 0; s >>= 1) {
            if (tid < s) red[tid] += red[tid + s];
            __syncthreads();
        }
        if (tid == 0) sscal = 1.f / red[0];
        __syncthreads();
        const float inv = sscal;

        const int d = tid & 63, g = tid >> 6;
        float accum = 0.f;
        for (int n = g; n < SEQ; n += 4) {
            const unsigned short* row = qkv + (size_t)(b * SEQ + n) * QKV_COLS + colOff;
            accum += sl[n] * bf2f(row[d]);
        }
        __syncthreads();
        red[tid] = accum; __syncthreads();
        if (tid < 64) {
            float gval = (red[tid] + red[tid + 64] + red[tid + 128] + red[tid + 192]) * inv;
            sg[tid] = gval;
            if (pass == 0) gq_out[bh * 64 + tid] = gval;
            else           gk_out[bh * 64 + tid] = gval;
        }
        __syncthreads();
    }
}

// ---------------------------------------------------------------------------
// r = (v * global_k) @ W_r^T + b_r + q, written as bf16 in [token, h*64+e].
// Block: one (b,h) x 128-token tile. K=64 -> 2 WMMA K-steps.
// ---------------------------------------------------------------------------
__global__ __launch_bounds__(256) void r_kernel(
    const unsigned short* __restrict__ qkv,
    const float* __restrict__ gk,
    const float* __restrict__ W_r, const float* __restrict__ b_r,
    unsigned short* __restrict__ r_out)
{
    constexpr int LDU = 72;   // 144B rows; frag offsets stay 16B-aligned
    __shared__ __align__(16) unsigned short sU[128 * LDU];  // u tile [token][d]
    __shared__ __align__(16) unsigned short sW[64 * LDU];   // W_r^T  [e][d]
    __shared__ float sgk[64];

    const int tid  = threadIdx.x;
    const int lane = tid & 31;
    const int wave = tid >> 5;            // 8 waves x 16 rows = 128 tokens
    const int bh   = blockIdx.x;
    const int b    = bh >> 4, h = bh & 15;
    const int tok0 = blockIdx.y * 128;

    if (tid < 64) sgk[tid] = gk[bh * 64 + tid];
    for (int i = tid; i < 64 * 64; i += 256) {
        int e = i >> 6, d = i & 63;
        sW[e * LDU + d] = f2bf(W_r[e * 64 + d]);   // B[k=d][n=e] stored transposed
    }
    __syncthreads();
    for (int i = tid; i < 128 * 64; i += 256) {
        int r = i >> 6, d = i & 63;
        size_t row = (size_t)(b * SEQ + tok0 + r);
        float vv = bf2f(qkv[row * QKV_COLS + 2 * INNER + h * 64 + d]);
        sU[r * LDU + d] = f2bf(vv * sgk[d]);
    }
    __syncthreads();

    v8f z = {0.f,0.f,0.f,0.f,0.f,0.f,0.f,0.f};
    v8f acc[4];
    #pragma unroll
    for (int u = 0; u < 4; ++u) acc[u] = z;

    #pragma unroll
    for (int ks = 0; ks < 64; ks += 32) {
        v16bf a = load_fragA(sU, wave * 16, LDU, ks, lane);
        #pragma unroll
        for (int u = 0; u < 4; ++u) {
            v16bf bb = load_fragB(sW, u * 16, LDU, ks, lane);
            acc[u] = __builtin_amdgcn_wmma_f32_16x16x32_bf16(
                false, a, false, bb, (short)0, acc[u], false, false);
        }
    }

    const int cn = lane & 15;
    const int rb = (lane >> 4) * 8;
    #pragma unroll
    for (int u = 0; u < 4; ++u) {
        int e = u * 16 + cn;
        float br = b_r[e];
        #pragma unroll
        for (int v = 0; v < 8; ++v) {
            int rloc = wave * 16 + rb + v;
            size_t row = (size_t)(b * SEQ + tok0 + rloc);
            float qv = bf2f(qkv[row * QKV_COLS + h * 64 + e]);
            r_out[row * (size_t)INNER + h * 64 + e] = f2bf(acc[u][v] + br + qv);
        }
    }
}

// ---------------------------------------------------------------------------
extern "C" void kernel_launch(void* const* d_in, const int* in_sizes, int n_in,
                              void* d_out, int out_size, void* d_ws, size_t ws_size,
                              hipStream_t stream)
{
    (void)in_sizes; (void)n_in; (void)out_size; (void)ws_size;
    const float* x    = (const float*)d_in[0];
    const float* Wqkv = (const float*)d_in[1];
    const float* wq   = (const float*)d_in[2];
    const float* wk   = (const float*)d_in[3];
    const float* W_r  = (const float*)d_in[4];
    const float* b_r  = (const float*)d_in[5];
    const float* Wout = (const float*)d_in[6];
    const float* bout = (const float*)d_in[7];
    float* out = (float*)d_out;

    unsigned short* x_bf   = (unsigned short*)d_ws;                     // 16384*1024
    unsigned short* Wqkv_t = x_bf   + (size_t)TOKENS * INPUT_DIM;       // 3072*1024 [N][K]
    unsigned short* Wout_t = Wqkv_t + (size_t)QKV_COLS * INPUT_DIM;     // 1024*1024 [N][K]
    unsigned short* qkv_ws = Wout_t + (size_t)INNER * INNER;            // 16384*3072
    unsigned short* r_ws   = qkv_ws + (size_t)TOKENS * QKV_COLS;        // 16384*1024
    float* gq_ws = (float*)(r_ws + (size_t)TOKENS * INNER);             // 64*64
    float* gk_ws = gq_ws + 64 * 64;

    // prep: bf16 conversion (x) and bf16 transposes (weights), done once
    convert_bf16_kernel<<<(TOKENS * INPUT_DIM / 4 + 255) / 256, 256, 0, stream>>>(
        x, x_bf, TOKENS * INPUT_DIM / 4);
    transpose_bf16_kernel<<<dim3(QKV_COLS / 64, INPUT_DIM / 64), 256, 0, stream>>>(
        Wqkv, Wqkv_t, INPUT_DIM, QKV_COLS);
    transpose_bf16_kernel<<<dim3(INNER / 64, INNER / 64), 256, 0, stream>>>(
        Wout, Wout_t, INNER, INNER);

    // qkv = x @ W_qkv  (bf16 out)
    gemm_wmma_async<true><<<dim3(QKV_COLS / 128, TOKENS / 128), 256, 0, stream>>>(
        x_bf, Wqkv_t, nullptr, qkv_ws, TOKENS, QKV_COLS, INPUT_DIM);

    global_qk_kernel<<<dim3(BATCH * HEADS), 256, 0, stream>>>(
        qkv_ws, wq, wk, gq_ws, gk_ws);

    r_kernel<<<dim3(BATCH * HEADS, SEQ / 128), 256, 0, stream>>>(
        qkv_ws, gk_ws, W_r, b_r, r_ws);

    // out = r @ W_out + b_out  (f32 out)
    gemm_wmma_async<false><<<dim3(INNER / 128, TOKENS / 128), 256, 0, stream>>>(
        r_ws, Wout_t, bout, out, TOKENS, INNER, INNER);
}
